// EquivariancePermutationLayer_87600152969226
// MI455X (gfx1250) — matrix operand order repaired and checked by reading, not compiled, not run on hardware
//
#include <hip/hip_runtime.h>
#include <hip/hip_bf16.h>

// ---------------------------------------------------------------------------
// Fused equivariance-MLP pipeline for MI455X (gfx1250, wave32, WMMA).
//
// Every MLP is  n_in -> 128 -> 128 -> 128 -> 1  (ReLU on first 3 layers).
// One wave computes a 32-row tile (two 16-row sub-tiles, register-blocked on M
// so each B fragment feeds two v_wmma_f32_16x16x32_f16 — halves LDS B traffic):
//   layer1: K padded to 32, bias folded in as constant-1 column -> 16 WMMAs
//   layer2: 128x128                                             -> 64 WMMAs
//   layer3: 128x128                                             -> 64 WMMAs
//   head  : 128->1 (B = col 0 only)                             ->  8 WMMAs
// Weights are pre-transposed to f16 [out][in] with rows padded to 136 halves
// (row stride = 272B = 4 banks mod 64 -> conflict-free fragment loads) and
// staged into LDS once per block.
// ---------------------------------------------------------------------------

typedef __attribute__((ext_vector_type(16))) _Float16 v16h;
typedef __attribute__((ext_vector_type(8)))  _Float16 v8h;
typedef __attribute__((ext_vector_type(8)))  float    v8f;
typedef __attribute__((ext_vector_type(4)))  unsigned int v4u;

// f16 weight block layout (bytes), one block per MLP:
constexpr int OFF_W1T = 0;        // [128][40]  f16  (layer1^T, K padded, bias in row n_in)
constexpr int OFF_W2T = 10240;    // [128][136] f16  (layer2^T, padded)
constexpr int OFF_W3T = 45056;    // [128][136] f16  (layer3^T, padded)
constexpr int OFF_W4  = 79872;    // [128]      f16  (head weights)
constexpr int OFF_B1  = 80128;    // [128] f32 (unused at runtime; kept for layout)
constexpr int OFF_B2  = 80640;    // [128] f32
constexpr int OFF_B3  = 81152;    // [128] f32
constexpr int OFF_B4  = 81664;    // f32
constexpr int WBLK    = 81920;    // block stride (80 KB)

constexpr int ACTP = 136;         // activation row pitch in halves (128 + 8 pad)
constexpr int W1P  = 40;          // layer1 weight row pitch in halves (32 + 8 pad)
constexpr int MROWS = 32;         // rows per wave tile (2 x 16 sub-tiles)
constexpr int ACT_BYTES = MROWS * ACTP * 2;                 // 8704 B per wave
constexpr int SMEM_BYTES = WBLK + 8 * ACT_BYTES;            // 151552 B (2 blocks/WGP)

__device__ __forceinline__ v16h cat16(v8h a, v8h b) {
  return __builtin_shufflevector(a, b, 0,1,2,3,4,5,6,7,8,9,10,11,12,13,14,15);
}

// One dense layer over a 32-row tile: act[32][K] x Wt[128][K] -> relu(+bias).
// A fragment (16x32 f16): lane l = row l&15; halves 0..7 = K hi*8.., 8..15 = K 16+hi*8..
// B fragment (32x16 f16): lane l = col l&15; halves 0..7 = K hi*8.., 8..15 = K 16+hi*8..
// D (16x16 f32): lane = col, VGPR r = row r+8*hi.
// All loads of the layer precede all stores -> in-place act update is safe
// (same-wave LDS ops are processed in order).
template<int KSTEPS, int WPITCH, bool BIAS>
__device__ __forceinline__ void mlp_layer(const _Float16* asrc, const _Float16* wt,
                                          const float* bias, _Float16* adst, int lane)
{
  const int nl = lane & 15;
  const int hi = lane >> 4;
  v8f c0[8], c1[8];
  #pragma unroll
  for (int nt = 0; nt < 8; ++nt)
    #pragma unroll
    for (int r = 0; r < 8; ++r) { c0[nt][r] = 0.0f; c1[nt][r] = 0.0f; }

  #pragma unroll
  for (int k = 0; k < KSTEPS; ++k) {
    const int k0 = k * 32;
    const _Float16* pa0 = asrc + nl * ACTP + k0 + hi * 8;
    const _Float16* pa1 = pa0 + 16 * ACTP;
    v16h A0 = cat16(*(const v8h*)pa0, *(const v8h*)(pa0 + 16));
    v16h A1 = cat16(*(const v8h*)pa1, *(const v8h*)(pa1 + 16));
    #pragma unroll
    for (int nt = 0; nt < 8; ++nt) {
      const _Float16* pb = wt + (nt * 16 + nl) * WPITCH + k0 + hi * 8;
      v16h Bm = cat16(*(const v8h*)pb, *(const v8h*)(pb + 16));
      c0[nt] = __builtin_amdgcn_wmma_f32_16x16x32_f16(false, A0, false, Bm,
                                                      (short)0, c0[nt], false, false);
      c1[nt] = __builtin_amdgcn_wmma_f32_16x16x32_f16(false, A1, false, Bm,
                                                      (short)0, c1[nt], false, false);
    }
  }
  asm volatile("" ::: "memory");   // keep LDS stores below the loads above
  #pragma unroll
  for (int nt = 0; nt < 8; ++nt) {
    const int n = nt * 16 + nl;
    const float bb = BIAS ? bias[n] : 0.0f;
    #pragma unroll
    for (int r = 0; r < 8; ++r) {
      float v0 = c0[nt][r];
      float v1 = c1[nt][r];
      if (BIAS) { v0 += bb; v1 += bb; }
      v0 = v0 > 0.0f ? v0 : 0.0f;
      v1 = v1 > 0.0f ? v1 : 0.0f;
      adst[(r + 8 * hi) * ACTP + n]        = (_Float16)v0;
      adst[(16 + r + 8 * hi) * ACTP + n]   = (_Float16)v1;
    }
  }
  asm volatile("" ::: "memory");   // same-wave LDS is in-order; just block reordering
}

// Head: 128 -> 1 dot product as WMMA with B = w4 in column 0 (other cols zero).
__device__ __forceinline__ void mlp_head(const _Float16* asrc, const _Float16* w4,
                                         float b4, float* __restrict__ out,
                                         int rowbase, int rows, int lane)
{
  const int nl = lane & 15;
  const int hi = lane >> 4;
  v8f c0, c1;
  #pragma unroll
  for (int r = 0; r < 8; ++r) { c0[r] = 0.0f; c1[r] = 0.0f; }
  #pragma unroll
  for (int k = 0; k < 4; ++k) {
    const int k0 = k * 32;
    const _Float16* pa0 = asrc + nl * ACTP + k0 + hi * 8;
    const _Float16* pa1 = pa0 + 16 * ACTP;
    v16h A0 = cat16(*(const v8h*)pa0, *(const v8h*)(pa0 + 16));
    v16h A1 = cat16(*(const v8h*)pa1, *(const v8h*)(pa1 + 16));
    v16h Bm;
    #pragma unroll
    for (int j = 0; j < 16; ++j) Bm[j] = (_Float16)0.0f;
    if (nl == 0) {
      const _Float16* pw = w4 + k0 + hi * 8;
      Bm = cat16(*(const v8h*)pw, *(const v8h*)(pw + 16));
    }
    c0 = __builtin_amdgcn_wmma_f32_16x16x32_f16(false, A0, false, Bm,
                                                (short)0, c0, false, false);
    c1 = __builtin_amdgcn_wmma_f32_16x16x32_f16(false, A1, false, Bm,
                                                (short)0, c1, false, false);
  }
  if (nl == 0) {                     // D column N==0 lives in lanes 0 and 16
    #pragma unroll
    for (int r = 0; r < 8; ++r) {
      const int row0 = rowbase + r + 8 * hi;
      const int row1 = rowbase + 16 + r + 8 * hi;
      if (row0 < rows) out[row0] = c0[r] + b4;
      if (row1 < rows) out[row1] = c1[r] + b4;
    }
  }
}

// Input-build modes (a constant-1 column at k = n_in feeds the folded L1 bias):
//  0: n_in=1 : scalars[b*45 + (off+i)*3 + comp],         row = b*cnt + i
//  1: n_in=15: scalars[row*45 + k*3 + 2], k<15
//  2: n_in=5 : scalars[row*45 + k*3 + 0], k<5
//  3: n_in=2 : [scalars[b,off+i,2], in_a[row](+in_b[row])]
//  4: n_in=4 : [scalars[b,off+i,2], in_a(+in_b), bc_r[b], bc_m[b]]
__global__ __launch_bounds__(256) void mlp_fwd_kernel(
    const float* __restrict__ scalars,
    const float* __restrict__ in_a, const float* __restrict__ in_b,
    const float* __restrict__ bc_r, const float* __restrict__ bc_m,
    const unsigned char* __restrict__ wblk,
    float* __restrict__ out,
    int rows, int mode, int cnt, int off, int comp, int tpw)
{
  extern __shared__ unsigned char smem[];

  // Warm the weight block into cache (global_prefetch_b8), then stage to LDS.
  for (int i = threadIdx.x; i < WBLK / 128; i += blockDim.x)
    __builtin_prefetch(wblk + i * 128, 0, 3);
  {
    const v4u* src = (const v4u*)wblk;
    v4u* dst = (v4u*)smem;
    for (int i = threadIdx.x; i < WBLK / 16; i += blockDim.x) dst[i] = src[i];
  }
  __syncthreads();

  const _Float16* w1t = (const _Float16*)(smem + OFF_W1T);
  const _Float16* w2t = (const _Float16*)(smem + OFF_W2T);
  const _Float16* w3t = (const _Float16*)(smem + OFF_W3T);
  const _Float16* w4  = (const _Float16*)(smem + OFF_W4);
  const float* b2 = (const float*)(smem + OFF_B2);
  const float* b3 = (const float*)(smem + OFF_B3);
  const float  b4 = *(const float*)(smem + OFF_B4);

  const int wave = threadIdx.x >> 5;
  const int lane = threadIdx.x & 31;
  _Float16* act = (_Float16*)(smem + WBLK) + wave * (MROWS * ACTP);

  for (int t = 0; t < tpw; ++t) {
    const int tile = (blockIdx.x * 8 + wave) * tpw + t;   // wave-uniform
    const int rowbase = tile * MROWS;
    if (rowbase >= rows) break;                            // uniform branch: EXEC stays full

    {   // all 32 lanes: one input row each
      const int row = rowbase + lane;
      _Float16* ar = act + lane * ACTP;
      const v8h z8 = {};
      #pragma unroll
      for (int q = 0; q < 4; ++q) *(v8h*)(ar + q * 8) = z8;   // 4x ds_store_b128
      if (row < rows) {
        if (mode == 0) {
          const int b = row / cnt, i = row - b * cnt;
          ar[0] = (_Float16)scalars[b * 45 + (off + i) * 3 + comp];
          ar[1] = (_Float16)1.0f;
        } else if (mode == 1) {
          for (int k = 0; k < 15; ++k) ar[k] = (_Float16)scalars[row * 45 + k * 3 + 2];
          ar[15] = (_Float16)1.0f;
        } else if (mode == 2) {
          for (int k = 0; k < 5; ++k) ar[k] = (_Float16)scalars[row * 45 + k * 3];
          ar[5] = (_Float16)1.0f;
        } else {
          const int b = row / cnt, i = row - b * cnt;
          ar[0] = (_Float16)scalars[b * 45 + (off + i) * 3 + 2];
          float s = in_a[row];
          if (in_b) s += in_b[row];
          ar[1] = (_Float16)s;
          if (mode == 4) {
            ar[2] = (_Float16)bc_r[b];
            ar[3] = (_Float16)bc_m[b];
            ar[4] = (_Float16)1.0f;
          } else {
            ar[2] = (_Float16)1.0f;
          }
        }
      }
    }
    asm volatile("" ::: "memory");

    mlp_layer<1, W1P,  false>(act, w1t, nullptr, act, lane); // 16 WMMA, bias folded
    mlp_layer<4, ACTP, true >(act, w2t, b2, act, lane);      // 64 WMMA
    mlp_layer<4, ACTP, true >(act, w3t, b3, act, lane);      // 64 WMMA
    mlp_head(act, w4, b4, out, rowbase, rows, lane);         //  8 WMMA
  }
}

// fp32 params -> f16 transposed + padded weight block.
// Layer-1 bias is folded into W1^T at row k = n_in (matching the constant-1
// input column), so layer 1 needs no runtime bias add.
__global__ __launch_bounds__(256) void prep_weights_kernel(
    const float* __restrict__ W0, const float* __restrict__ B0,
    const float* __restrict__ W1, const float* __restrict__ B1,
    const float* __restrict__ W2, const float* __restrict__ B2,
    const float* __restrict__ W3, const float* __restrict__ B3,
    int n_in, unsigned char* __restrict__ dstblk)
{
  _Float16* w1t = (_Float16*)(dstblk + OFF_W1T);
  _Float16* w2t = (_Float16*)(dstblk + OFF_W2T);
  _Float16* w3t = (_Float16*)(dstblk + OFF_W3T);
  _Float16* w4  = (_Float16*)(dstblk + OFF_W4);
  float* b1 = (float*)(dstblk + OFF_B1);
  float* b2 = (float*)(dstblk + OFF_B2);
  float* b3 = (float*)(dstblk + OFF_B3);
  float* b4 = (float*)(dstblk + OFF_B4);
  const int tid = blockIdx.x * blockDim.x + threadIdx.x;
  const int np  = gridDim.x * blockDim.x;
  for (int i = tid; i < 128 * W1P; i += np) {
    const int n = i / W1P, k = i - n * W1P;
    float v = 0.0f;
    if (k < n_in)       v = W0[k * 128 + n];
    else if (k == n_in) v = B0[n];               // folded layer-1 bias
    w1t[i] = (_Float16)v;
  }
  for (int i = tid; i < 128 * ACTP; i += np) {
    const int n = i / ACTP, k = i - n * ACTP;
    const _Float16 z = (_Float16)0.0f;
    w2t[i] = (k < 128) ? (_Float16)W1[k * 128 + n] : z;
    w3t[i] = (k < 128) ? (_Float16)W2[k * 128 + n] : z;
  }
  for (int i = tid; i < 128; i += np) {
    w4[i] = (_Float16)W3[i];
    b1[i] = B0[i]; b2[i] = B1[i]; b3[i] = B2[i];
  }
  if (tid == 0) b4[0] = B3[0];
}

// Final einsum/combine: out[b][3x3] = sI*x[b,15] + sum_i uMij*x[b,5+i] + sum_i uMii*x[b,i]
__global__ __launch_bounds__(256) void combine_kernel(
    const float* __restrict__ x,
    const float* __restrict__ uIij, const float* __restrict__ uIii,
    const float* __restrict__ uMij, const float* __restrict__ uMii,
    float* __restrict__ out, int Bn)
{
  const int b = blockIdx.x * blockDim.x + threadIdx.x;
  if (b >= Bn) return;
  float sI = 0.0f;
  #pragma unroll
  for (int i = 0; i < 10; ++i) sI += uIij[b * 10 + i];
  #pragma unroll
  for (int i = 0; i < 5; ++i)  sI += uIii[b * 5 + i];
  const float* xb = x + (size_t)b * 144;
  float o[9];
  #pragma unroll
  for (int j = 0; j < 9; ++j) o[j] = sI * xb[15 * 9 + j];
  #pragma unroll
  for (int i = 0; i < 10; ++i) {
    const float w = uMij[b * 10 + i];
    #pragma unroll
    for (int j = 0; j < 9; ++j) o[j] += w * xb[(5 + i) * 9 + j];
  }
  #pragma unroll
  for (int i = 0; i < 5; ++i) {
    const float w = uMii[b * 5 + i];
    #pragma unroll
    for (int j = 0; j < 9; ++j) o[j] += w * xb[i * 9 + j];
  }
  #pragma unroll
  for (int j = 0; j < 9; ++j) out[(size_t)b * 9 + j] = o[j];
}

extern "C" void kernel_launch(void* const* d_in, const int* in_sizes, int n_in_cnt,
                              void* d_out, int out_size, void* d_ws, size_t ws_size,
                              hipStream_t stream) {
  const float* scalars = (const float*)d_in[0];
  const float* x       = (const float*)d_in[1];
  const int Bn = in_sizes[0] / 45;
  unsigned char* ws = (unsigned char*)d_ws;

  // params in setup_inputs() dict (insertion) order:
  // f_Mij, f_Mii, f_Mrij, f_Mrii, f_Mmij, f_Mmii, f_Mmimj, f_Mmimi, f_Iij, f_Iii, f_Imij, f_Imii
  static const int NIN[12] = {4, 4, 15, 15, 5, 5, 1, 1, 2, 2, 1, 1};
  auto P = [&](int m, int j) { return (const float*)d_in[2 + m * 8 + j]; };
  auto wptr = [&](int m) { return ws + (size_t)m * WBLK; };

  for (int m = 0; m < 12; ++m)
    prep_weights_kernel<<<64, 256, 0, stream>>>(
        P(m,0), P(m,1), P(m,2), P(m,3), P(m,4), P(m,5), P(m,6), P(m,7),
        NIN[m], wptr(m));

  // scratch layout after the 12 weight blocks
  size_t o = (size_t)12 * WBLK;
  auto alloc = [&](size_t n) { float* p = (float*)(ws + o); o += n * 4; return p; };
  float* tIa  = alloc((size_t)Bn * 10);
  float* tIb  = alloc((size_t)Bn * 10);
  float* tIi  = alloc((size_t)Bn * 5);
  float* tMa  = alloc((size_t)Bn * 10);
  float* tMb  = alloc((size_t)Bn * 10);
  float* tMi  = alloc((size_t)Bn * 5);
  float* rij  = alloc((size_t)Bn);
  float* rii  = alloc((size_t)Bn);
  float* mij  = alloc((size_t)Bn);
  float* mii  = alloc((size_t)Bn);
  float* uIij = alloc((size_t)Bn * 10);
  float* uIii = alloc((size_t)Bn * 5);
  float* uMij = alloc((size_t)Bn * 10);
  float* uMii = alloc((size_t)Bn * 5);
  (void)ws_size;

  (void)hipFuncSetAttribute((const void*)mlp_fwd_kernel,
                            hipFuncAttributeMaxDynamicSharedMemorySize, SMEM_BYTES);

  auto launch_mlp = [&](int rows, int mode, int cnt, int off, int comp, int widx,
                        const float* ina, const float* inb,
                        const float* bcr, const float* bcm, float* outp) {
    // Small launches: 1 tile/wave for max blocks; big ones: 2 (amortize staging).
    const int tpw   = (rows > 262144) ? 2 : 1;
    const int tiles  = (rows + MROWS - 1) / MROWS;
    const int blocks = (tiles + 8 * tpw - 1) / (8 * tpw);
    mlp_fwd_kernel<<<blocks, 256, SMEM_BYTES, stream>>>(
        scalars, ina, inb, bcr, bcm, wptr(widx), outp, rows, mode, cnt, off, comp, tpw);
  };

  // ---- stage 1: MLPs on raw scalars ----
  launch_mlp(Bn * 10, 0, 10, 5, 0, 10, nullptr, nullptr, nullptr, nullptr, tIa); // f_Imij(sij0)
  launch_mlp(Bn * 10, 0, 10, 5, 1, 10, nullptr, nullptr, nullptr, nullptr, tIb); // f_Imij(sij1)
  launch_mlp(Bn * 5,  0, 5,  0, 0, 11, nullptr, nullptr, nullptr, nullptr, tIi); // f_Imii(sii0)
  launch_mlp(Bn * 10, 0, 10, 5, 0, 6,  nullptr, nullptr, nullptr, nullptr, tMa); // f_Mmimj(sij0)
  launch_mlp(Bn * 10, 0, 10, 5, 1, 6,  nullptr, nullptr, nullptr, nullptr, tMb); // f_Mmimj(sij1)
  launch_mlp(Bn * 5,  0, 5,  0, 0, 7,  nullptr, nullptr, nullptr, nullptr, tMi); // f_Mmimi(sii0)
  launch_mlp(Bn,      1, 1,  0, 0, 2,  nullptr, nullptr, nullptr, nullptr, rij); // f_Mrij
  launch_mlp(Bn,      1, 1,  0, 0, 3,  nullptr, nullptr, nullptr, nullptr, rii); // f_Mrii
  launch_mlp(Bn,      2, 1,  0, 0, 4,  nullptr, nullptr, nullptr, nullptr, mij); // f_Mmij
  launch_mlp(Bn,      2, 1,  0, 0, 5,  nullptr, nullptr, nullptr, nullptr, mii); // f_Mmii

  // ---- stage 2: MLPs on concatenated features ----
  launch_mlp(Bn * 10, 3, 10, 5, 0, 8, tIa, tIb, nullptr, nullptr, uIij);         // f_Iij
  launch_mlp(Bn * 5,  3, 5,  0, 0, 9, tIi, nullptr, nullptr, nullptr, uIii);     // f_Iii
  launch_mlp(Bn * 10, 4, 10, 5, 0, 0, tMa, tMb, rij, mij, uMij);                 // f_Mij
  launch_mlp(Bn * 5,  4, 5,  0, 0, 1, tMi, nullptr, rii, mii, uMii);             // f_Mii

  // ---- stage 3: combine with x ----
  combine_kernel<<<(Bn + 255) / 256, 256, 0, stream>>>(
      x, uIij, uIii, uMij, uMii, (float*)d_out, Bn);
  (void)out_size; (void)n_in_cnt;
}
